// DenoisePretrainModel_89799176225200
// MI455X (gfx1250) — matrix operand order repaired and checked by reading, not compiled, not run on hardware
//
#include <hip/hip_runtime.h>

typedef unsigned short u16;
typedef unsigned int   u32;

typedef __bf16 bf16;
typedef bf16  v16bf __attribute__((ext_vector_type(16)));
typedef float v8f   __attribute__((ext_vector_type(8)));

#define GG   8
#define NBG  1024
#define APB  4
#define HID  512
#define KNN  9
#define NBB  (GG*NBG)          // 8192 blocks
#define NUU  (NBB*APB)         // 32768 atoms
#define EHALF (NBB*KNN)        // 73728 edges per half
#define BIGV 1.0e9f

// d_out float offsets
#define OFF_E     0
#define OFF_NOISE 8
#define OFF_UNIT  (8 + NUU*3)
#define OFF_BLOCK (OFF_UNIT + NUU*HID)
#define OFF_GRAPH (OFF_BLOCK + NBB*HID)
#define OFF_LOSS  (OFF_GRAPH + GG*HID)

__device__ __forceinline__ float siluf(float x){ return x / (1.0f + __expf(-x)); }
__device__ __forceinline__ u16 f2bf(float f){
  u32 x = __float_as_uint(f);
  u32 r = x + 0x7fffu + ((x >> 16) & 1u);
  return (u16)(r >> 16);
}
__device__ __forceinline__ float bf2f(u16 u){ return __uint_as_float(((u32)u) << 16); }

union FragH { v16bf v; uint4 q[2]; };
union FragC { v8f  v; float f[8]; };

// ---------------- prep kernels ----------------
__global__ void k_zero(float* out, float* nl){
  int t = threadIdx.x;
  if (t < 8) out[OFF_E + t] = 0.0f;
  else if (t < 16) nl[t-8] = 0.0f;
}

__global__ void k_prep_w(const float* __restrict__ Wm, const float* __restrict__ Wu,
                         const float* __restrict__ W1,
                         u16* __restrict__ WmT, u16* __restrict__ WuT, u16* __restrict__ W1T){
  int idx = blockIdx.x*256 + threadIdx.x;
  if (idx >= 3*HID*HID) return;
  int l = idx >> 18; int r = idx & (HID*HID-1);
  int k = r >> 9; int n = r & 511;
  int to = l*HID*HID + n*HID + k;
  WmT[to] = f2bf(Wm[idx]);
  WuT[to] = f2bf(Wu[idx]);
  if (l == 0) W1T[n*HID + k] = f2bf(W1[r]);
}

__global__ void k_ebias(const float* __restrict__ edge_emb, const float* __restrict__ We,
                        float* __restrict__ ebias){
  int idx = blockIdx.x*256 + threadIdx.x;
  if (idx >= 3*2*HID) return;
  int l = idx >> 10; int t = (idx >> 9) & 1; int n = idx & 511;
  float s = 0.0f;
  for (int k = 0; k < 64; ++k)
    s += edge_emb[t*64 + k] * We[l*64*HID + k*HID + n];
  ebias[idx] = s;
}

__global__ void k_perturb(const float* __restrict__ Z, const float* __restrict__ noise,
                          const int* __restrict__ nlevel, const float* __restrict__ sigmas,
                          float* __restrict__ Zp){
  int idx = blockIdx.x*256 + threadIdx.x;
  if (idx >= NUU*3) return;
  int g = idx / (NBG*APB*3);
  float sig = sigmas[nlevel[g]];
  Zp[idx] = Z[idx] + noise[idx] * sig;
}

__global__ void k_zb0(const float* __restrict__ Zp, float* __restrict__ Zb0, float* __restrict__ ZbA){
  int idx = blockIdx.x*256 + threadIdx.x;
  if (idx >= NBB*3) return;
  int b = idx / 3; int d = idx % 3;
  float v = 0.25f*(Zp[(b*4+0)*3+d] + Zp[(b*4+1)*3+d] + Zp[(b*4+2)*3+d] + Zp[(b*4+3)*3+d]);
  Zb0[idx] = v; ZbA[idx] = v;
}

__global__ void k_h0(const int* __restrict__ A, const int* __restrict__ ap, const int* __restrict__ Bt,
                     const float* __restrict__ atom_emb, const float* __restrict__ pos_emb,
                     const float* __restrict__ block_emb, float* __restrict__ out){
  int idx = blockIdx.x*256 + threadIdx.x; // NU*512
  int u = idx >> 9; int c = idx & 511;
  out[OFF_UNIT + idx] = atom_emb[A[u]*HID + c] + pos_emb[ap[u]*HID + c] + block_emb[Bt[u>>2]*HID + c];
}

__global__ void k_hinit(const float* __restrict__ out, float* __restrict__ h, u16* __restrict__ hb){
  int idx = blockIdx.x*256 + threadIdx.x; // NB*512
  int b = idx >> 9; int c = idx & 511;
  const float* U = out + OFF_UNIT;
  float v = 0.25f*(U[(b*4+0)*HID+c] + U[(b*4+1)*HID+c] + U[(b*4+2)*HID+c] + U[(b*4+3)*HID+c]);
  h[idx] = v; hb[idx] = f2bf(v);
}

// ---------------- distance + knn ----------------
__global__ __launch_bounds__(256) void k_dist(const float* __restrict__ Zp, float* __restrict__ D){
  int tj = blockIdx.x, ti = blockIdx.y, g = blockIdx.z;
  __shared__ float si[192], sj[192];
  int basei = (g*NBG + ti*16)*APB;  // atom base
  int basej = (g*NBG + tj*16)*APB;
  for (int t = threadIdx.x; t < 192; t += 256){
    si[t] = Zp[basei*3 + t];
    sj[t] = Zp[basej*3 + t];
  }
  __syncthreads();
  int il = threadIdx.x >> 4, jl = threadIdx.x & 15;
  float best = 3.4e38f;
  #pragma unroll
  for (int a2 = 0; a2 < 4; ++a2){
    float x0 = si[(il*4+a2)*3+0], y0 = si[(il*4+a2)*3+1], z0 = si[(il*4+a2)*3+2];
    #pragma unroll
    for (int b2 = 0; b2 < 4; ++b2){
      float dx = x0 - sj[(jl*4+b2)*3+0];
      float dy = y0 - sj[(jl*4+b2)*3+1];
      float dz = z0 - sj[(jl*4+b2)*3+2];
      best = fminf(best, dx*dx + dy*dy + dz*dz);
    }
  }
  D[((size_t)(g*NBG + ti*16 + il))*NBG + tj*16 + jl] = best;
}

__global__ void k_knn(const float* __restrict__ D, const int* __restrict__ seg,
                      int* __restrict__ srcI, int* __restrict__ srcJ){
  int r = blockIdx.x;            // global block row 0..8191
  int g = r / NBG, i = r % NBG;
  int lane = threadIdx.x;        // 32 threads
  __shared__ float vals[NBG];
  const float* Drow = D + (size_t)r*NBG;
  int segi = seg[r];
  for (int phase = 0; phase < 2; ++phase){
    for (int t = lane; t < NBG; t += 32){
      int sj = seg[g*NBG + t];
      bool ok = phase == 0 ? (sj == segi && t != i) : (sj != segi);
      vals[t] = ok ? Drow[t] : BIGV;
    }
    __syncthreads();
    for (int k = 0; k < KNN; ++k){
      float best = 3.4e38f; int bi = lane;
      for (int t = lane; t < NBG; t += 32){
        float v = vals[t];
        if (v < best){ best = v; bi = t; }
      }
      for (int off = 16; off >= 1; off >>= 1){
        float ob = __shfl_xor(best, off);
        int   oi = __shfl_xor(bi, off);
        if (ob < best || (ob == best && oi < bi)){ best = ob; bi = oi; }
      }
      if (lane == 0){
        (phase ? srcJ : srcI)[r*KNN + k] = g*NBG + bi;
        vals[bi] = BIGV;
      }
      __syncthreads();
    }
    __syncthreads();
  }
}

// ---------------- WMMA GEMM: C = A[M x 512] @ B[512 x 512]  (BT is [n][k] bf16) ----------------
// EPI 0: outHm = C ; EPI 1: houtf = hold + silu(C), houtb = bf16(same)
// EPI 2: energy: atomicAdd per graph of sum( silu(C + b1[n]) * W2[n] )
template<int EPI>
__global__ __launch_bounds__(256) void gemm512(
    const u16* __restrict__ A, const u16* __restrict__ BT,
    float* __restrict__ outHm,
    const float* __restrict__ hold, float* __restrict__ houtf, u16* __restrict__ houtb,
    const float* __restrict__ b1, const float* __restrict__ W2, float* __restrict__ outE)
{
  int wave = threadIdx.x >> 5;
  int lane = threadIdx.x & 31;
  int laneHi = lane >> 4;
  int lm = lane & 15;
  int rowBase = blockIdx.x*64 + (wave & 1)*32;
  int colBase = blockIdx.y*256 + (wave >> 1)*64;

  FragC c[2][4];
  #pragma unroll
  for (int mi = 0; mi < 2; ++mi)
    #pragma unroll
    for (int ni = 0; ni < 4; ++ni)
      #pragma unroll
      for (int j = 0; j < 8; ++j) c[mi][ni].f[j] = 0.0f;

  for (int kt = 0; kt < 16; ++kt){
    int kbA = kt*32 + 8*laneHi;
    int kbB = kt*32 + 8*laneHi;
    if (kt < 15){
      __builtin_prefetch(A  + (size_t)(rowBase + lm)*HID + kbA + 32, 0, 3);
      __builtin_prefetch(BT + (size_t)(colBase + lm)*HID + kbB + 32, 0, 3);
    }
    FragH a[2];
    #pragma unroll
    for (int mi = 0; mi < 2; ++mi){
      const uint4* pa = (const uint4*)(A + (size_t)(rowBase + mi*16 + lm)*HID + kbA);
      a[mi].q[0] = pa[0];   // K offsets kbA..kbA+7
      a[mi].q[1] = pa[2];   // K offsets kbA+16..kbA+23
    }
    FragH b[4];
    #pragma unroll
    for (int ni = 0; ni < 4; ++ni){
      const uint4* pb = (const uint4*)(BT + (size_t)(colBase + ni*16 + lm)*HID + kbB);
      b[ni].q[0] = pb[0]; b[ni].q[1] = pb[2];
    }
    #pragma unroll
    for (int mi = 0; mi < 2; ++mi)
      #pragma unroll
      for (int ni = 0; ni < 4; ++ni)
        c[mi][ni].v = __builtin_amdgcn_wmma_f32_16x16x32_bf16(
            false, a[mi].v, false, b[ni].v, (short)0, c[mi][ni].v, false, false);
  }

  if (EPI == 0){
    #pragma unroll
    for (int mi = 0; mi < 2; ++mi)
      #pragma unroll
      for (int ni = 0; ni < 4; ++ni)
        #pragma unroll
        for (int r2 = 0; r2 < 8; ++r2){
          int row = rowBase + mi*16 + r2 + 8*laneHi;
          int col = colBase + ni*16 + lm;
          outHm[(size_t)row*HID + col] = c[mi][ni].f[r2];
        }
  } else if (EPI == 1){
    #pragma unroll
    for (int mi = 0; mi < 2; ++mi)
      #pragma unroll
      for (int ni = 0; ni < 4; ++ni)
        #pragma unroll
        for (int r2 = 0; r2 < 8; ++r2){
          int row = rowBase + mi*16 + r2 + 8*laneHi;
          int col = colBase + ni*16 + lm;
          size_t idx = (size_t)row*HID + col;
          float v = hold[idx] + siluf(c[mi][ni].f[r2]);
          houtf[idx] = v;
          houtb[idx] = f2bf(v);
        }
  } else {
    float tot = 0.0f;
    #pragma unroll
    for (int ni = 0; ni < 4; ++ni){
      int col = colBase + ni*16 + lm;
      float w2v = W2[col], b1v = b1[col];
      #pragma unroll
      for (int mi = 0; mi < 2; ++mi)
        #pragma unroll
        for (int r2 = 0; r2 < 8; ++r2)
          tot += siluf(c[mi][ni].f[r2] + b1v) * w2v;
    }
    for (int off = 16; off >= 1; off >>= 1) tot += __shfl_xor(tot, off);
    __shared__ float eg;
    if (threadIdx.x == 0) eg = 0.0f;
    __syncthreads();
    if ((threadIdx.x & 31) == 0) atomicAdd(&eg, tot);
    __syncthreads();
    if (threadIdx.x == 0) atomicAdd(&outE[OFF_E + ((blockIdx.x*64) >> 10)], eg);
  }
}

// ---------------- fused message / aggregate / coef ----------------
__global__ __launch_bounds__(256) void k_msg(
    const float* __restrict__ Hm, const int* __restrict__ srcI, const int* __restrict__ srcJ,
    const float* __restrict__ ebias /*[2][512]*/, const float* __restrict__ wz,
    u16* __restrict__ aggb, float* __restrict__ coefOut)
{
  int b = blockIdx.x;
  int t = threadIdx.x;
  __shared__ int   srcs[18];
  __shared__ float coefs[18];
  if (t < 9)       srcs[t] = srcI[b*KNN + t];
  else if (t < 18) srcs[t] = srcJ[b*KNN + t - 9];
  if (t < 18) coefs[t] = 0.0f;
  __syncthreads();

  int c0 = t, c1 = t + 256;
  float h0 = Hm[(size_t)b*HID + c0], h1 = Hm[(size_t)b*HID + c1];
  float e00 = ebias[c0],        e01 = ebias[c1];
  float e10 = ebias[HID + c0],  e11 = ebias[HID + c1];
  float w0 = wz[c0], w1 = wz[c1];
  float a0 = 0.0f, a1 = 0.0f;
  for (int j = 0; j < 18; ++j){
    int s = srcs[j];
    float eb0 = (j >= 9) ? e10 : e00;
    float eb1 = (j >= 9) ? e11 : e01;
    float m0 = siluf(Hm[(size_t)s*HID + c0] + h0 + eb0);
    float m1 = siluf(Hm[(size_t)s*HID + c1] + h1 + eb1);
    a0 += m0; a1 += m1;
    atomicAdd(&coefs[j], m0*w0 + m1*w1);
  }
  __syncthreads();
  aggb[(size_t)b*HID + c0] = f2bf(a0);
  aggb[(size_t)b*HID + c1] = f2bf(a1);
  if (t < 9)       coefOut[b*KNN + t] = tanhf(coefs[t]);
  else if (t < 18) coefOut[EHALF + b*KNN + (t - 9)] = tanhf(coefs[t]);
}

__global__ void k_zb(const float* __restrict__ Zc, float* __restrict__ Zn,
                     const int* __restrict__ srcI, const int* __restrict__ srcJ,
                     const float* __restrict__ coef){
  int b = blockIdx.x*256 + threadIdx.x;
  if (b >= NBB) return;
  float z0 = Zc[b*3+0], z1 = Zc[b*3+1], z2 = Zc[b*3+2];
  float a0 = 0.0f, a1 = 0.0f, a2 = 0.0f;
  for (int j = 0; j < 18; ++j){
    int s  = (j < 9) ? srcI[b*KNN + j] : srcJ[b*KNN + j - 9];
    float cf = (j < 9) ? coef[b*KNN + j] : coef[EHALF + b*KNN + j - 9];
    a0 += (Zc[s*3+0] - z0)*cf;
    a1 += (Zc[s*3+1] - z1)*cf;
    a2 += (Zc[s*3+2] - z2)*cf;
  }
  float inv = 1.0f / (2.0f*KNN);
  Zn[b*3+0] = z0 + a0*inv;
  Zn[b*3+1] = z1 + a1*inv;
  Zn[b*3+2] = z2 + a2*inv;
}

// ---------------- outputs ----------------
__global__ void k_sprep(const float* __restrict__ h, u16* __restrict__ sb){
  int idx = blockIdx.x*256 + threadIdx.x;
  sb[idx] = f2bf(siluf(h[idx]));
}
__global__ void k_bcopy(const float* __restrict__ h, float* __restrict__ out){
  int idx = blockIdx.x*256 + threadIdx.x;
  out[OFF_BLOCK + idx] = h[idx];
}
__global__ void k_gsum(const float* __restrict__ h, float* __restrict__ out){
  int idx = blockIdx.x*256 + threadIdx.x; // G*512
  if (idx >= GG*HID) return;
  int g = idx >> 9, c = idx & 511;
  float s = 0.0f;
  for (int b = 0; b < NBG; ++b) s += h[(size_t)(g*NBG + b)*HID + c];
  out[OFF_GRAPH + idx] = s;
}
__global__ void k_uadd(const float* __restrict__ h, float* __restrict__ out){
  int idx = blockIdx.x*256 + threadIdx.x; // NU*512
  int u = idx >> 9, c = idx & 511;
  out[OFF_UNIT + idx] += h[(size_t)(u >> 2)*HID + c];
}
__global__ void k_loss(const float* __restrict__ Zf, const float* __restrict__ Zb0,
                       const float* __restrict__ noise, float* __restrict__ nl,
                       float* __restrict__ out){
  int u = blockIdx.x*256 + threadIdx.x;
  if (u >= NUU) return;
  int b = u >> 2, g = u >> 12;
  float s = 0.0f;
  for (int d = 0; d < 3; ++d){
    float dz = Zf[b*3+d] - Zb0[b*3+d];
    out[OFF_NOISE + u*3 + d] = dz;
    float df = dz - noise[u*3 + d];
    s += df*df;
  }
  atomicAdd(&nl[g], s);
}
__global__ void k_lossfin(const float* __restrict__ nl, float* __restrict__ out){
  float s = 0.0f;
  for (int g = 0; g < GG; ++g) s += nl[g];
  out[OFF_LOSS] = 0.5f * s / (float)GG;
}

// ---------------- host ----------------
extern "C" void kernel_launch(void* const* d_in, const int* in_sizes, int n_in,
                              void* d_out, int out_size, void* d_ws, size_t ws_size,
                              hipStream_t stream) {
  const float* Z         = (const float*)d_in[0];
  const int*   Bt        = (const int*)  d_in[1];
  const int*   A         = (const int*)  d_in[2];
  const int*   ap        = (const int*)  d_in[3];
  const int*   seg       = (const int*)  d_in[6];
  const float* noise     = (const float*)d_in[7];
  const int*   nlevel    = (const int*)  d_in[8];
  const float* sigmas    = (const float*)d_in[9];
  const float* atom_emb  = (const float*)d_in[10];
  const float* pos_emb   = (const float*)d_in[11];
  const float* block_emb = (const float*)d_in[12];
  const float* edge_emb  = (const float*)d_in[13];
  const float* Wm        = (const float*)d_in[14];
  const float* We        = (const float*)d_in[15];
  const float* Wu        = (const float*)d_in[16];
  const float* wz        = (const float*)d_in[17];
  const float* W1        = (const float*)d_in[18];
  const float* b1        = (const float*)d_in[19];
  const float* W2        = (const float*)d_in[20];
  float* out = (float*)d_out;

  char* ws = (char*)d_ws;
  size_t o = 0;
  auto alloc = [&](size_t bytes)->size_t { size_t r = o; o = (o + bytes + 255) & ~(size_t)255; return r; };
  float* Zp   = (float*)(ws + alloc((size_t)NUU*3*4));
  float* Zb0  = (float*)(ws + alloc((size_t)NBB*3*4));
  float* ZbA  = (float*)(ws + alloc((size_t)NBB*3*4));
  float* ZbB  = (float*)(ws + alloc((size_t)NBB*3*4));
  float* hA   = (float*)(ws + alloc((size_t)NBB*HID*4));
  float* hB   = (float*)(ws + alloc((size_t)NBB*HID*4));
  u16*   hb   = (u16*)  (ws + alloc((size_t)NBB*HID*2));
  float* coef = (float*)(ws + alloc((size_t)2*EHALF*4));
  int*   srcI = (int*)  (ws + alloc((size_t)EHALF*4));
  int*   srcJ = (int*)  (ws + alloc((size_t)EHALF*4));
  u16*   WmT  = (u16*)  (ws + alloc((size_t)3*HID*HID*2));
  u16*   WuT  = (u16*)  (ws + alloc((size_t)3*HID*HID*2));
  u16*   W1T  = (u16*)  (ws + alloc((size_t)HID*HID*2));
  float* ebias= (float*)(ws + alloc((size_t)3*2*HID*4));
  float* nl   = (float*)(ws + alloc(64));
  // big region: D overlaps (Hm | aggb | sb); D is consumed before layers start
  size_t big  = alloc((size_t)GG*NBG*NBG*4);   // 33.5 MB
  float* Dm   = (float*)(ws + big);
  float* Hm   = (float*)(ws + big);
  u16*   aggb = (u16*)  (ws + big + (size_t)NBB*HID*4);
  u16*   sb   = (u16*)  (ws + big + (size_t)NBB*HID*4 + (size_t)NBB*HID*2);

  k_zero<<<1, 32, 0, stream>>>(out, nl);
  k_prep_w<<<(3*HID*HID + 255)/256, 256, 0, stream>>>(Wm, Wu, W1, WmT, WuT, W1T);
  k_ebias<<<(3*2*HID + 255)/256, 256, 0, stream>>>(edge_emb, We, ebias);
  k_perturb<<<(NUU*3 + 255)/256, 256, 0, stream>>>(Z, noise, nlevel, sigmas, Zp);
  k_zb0<<<(NBB*3 + 255)/256, 256, 0, stream>>>(Zp, Zb0, ZbA);
  k_h0<<<(NUU*HID)/256, 256, 0, stream>>>(A, ap, Bt, atom_emb, pos_emb, block_emb, out);
  k_hinit<<<(NBB*HID)/256, 256, 0, stream>>>(out, hA, hb);
  k_dist<<<dim3(64, 64, 8), 256, 0, stream>>>(Zp, Dm);
  k_knn<<<NBB, 32, 0, stream>>>(Dm, seg, srcI, srcJ);

  float* hc = hA; float* hn = hB;
  float* zc = ZbA; float* zn = ZbB;
  for (int l = 0; l < 3; ++l){
    gemm512<0><<<dim3(128, 2), 256, 0, stream>>>(hb, WmT + (size_t)l*HID*HID, Hm,
                                                 nullptr, nullptr, nullptr, nullptr, nullptr, nullptr);
    k_msg<<<NBB, 256, 0, stream>>>(Hm, srcI, srcJ, ebias + l*2*HID, wz + l*HID, aggb, coef);
    gemm512<1><<<dim3(128, 2), 256, 0, stream>>>(aggb, WuT + (size_t)l*HID*HID, nullptr,
                                                 hc, hn, hb, nullptr, nullptr, nullptr);
    k_zb<<<(NBB + 255)/256, 256, 0, stream>>>(zc, zn, srcI, srcJ, coef);
    float* tf = hc; hc = hn; hn = tf;
    float* tz = zc; zc = zn; zn = tz;
  }
  float* hfin = hc;   // after 3 swaps
  float* zfin = zc;

  k_sprep<<<(NBB*HID)/256, 256, 0, stream>>>(hfin, sb);
  gemm512<2><<<dim3(128, 2), 256, 0, stream>>>(sb, W1T, nullptr,
                                               nullptr, nullptr, nullptr, b1, W2, out);
  k_bcopy<<<(NBB*HID)/256, 256, 0, stream>>>(hfin, out);
  k_gsum<<<(GG*HID + 255)/256, 256, 0, stream>>>(hfin, out);
  k_uadd<<<(NUU*HID)/256, 256, 0, stream>>>(hfin, out);
  k_loss<<<(NUU + 255)/256, 256, 0, stream>>>(zfin, Zb0, noise, nl, out);
  k_lossfin<<<1, 1, 0, stream>>>(nl, out);
  (void)in_sizes; (void)n_in; (void)out_size; (void)ws_size;
}
